// LearnedRewardModel_36352603193533
// MI455X (gfx1250) — matrix operand order repaired and checked by reading, not compiled, not run on hardware
//
#include <hip/hip_runtime.h>

typedef __attribute__((ext_vector_type(16))) _Float16 v16h;
typedef __attribute__((ext_vector_type(8)))  float    v8f;

#define TT        20
#define EE        64
#define HH        128
#define NG        512          // 4*H
#define KC        192          // H + E (concatenated K for fused gate GEMM)
#define WAVES     4
#define THREADS   (WAVES * 32)           // 128
#define BLK_ROWS  (WAVES * 16)           // 64 rows per block -> 256 blocks

// LDS layout (bytes)
#define OFF_WC    0                      // [512][192] f16  = 196608
#define OFF_W1    196608                 // [64][128]  f16  = 16384
#define OFF_BIAS  212992                 // [512]      f32  = 2048
#define OFF_AS    215040                 // [WAVES][16][192] f16
#define SMEM_BYTES (OFF_AS + WAVES * 16 * KC * 2)   // 239616

struct alignas(16) C16 { unsigned v[4]; };
struct alignas(16) F4  { float x, y, z, w; };
struct C32 { C16 lo, hi; };

__device__ __forceinline__ v16h frag_load(const _Float16* p0, const _Float16* p1) {
  C32 u;
  u.lo = *(const C16*)p0;
  u.hi = *(const C16*)p1;
  return __builtin_bit_cast(v16h, u);
}

__device__ __forceinline__ v8f wmma_f16(v16h a, v16h b, v8f c) {
  // D = A*B + C, f32 accumulate  ->  v_wmma_f32_16x16x32_f16
  return __builtin_amdgcn_wmma_f32_16x16x32_f16(false, a, false, b, (short)0, c, false, false);
}

__device__ __forceinline__ float frcp_(float x) { return __builtin_amdgcn_rcpf(x); }
__device__ __forceinline__ float fexp_(float x) {
  return __builtin_amdgcn_exp2f(x * 1.44269504088896340736f);
}
__device__ __forceinline__ float fsig_(float x) { return frcp_(1.0f + fexp_(-x)); }
__device__ __forceinline__ float ftanh_(float x) {
#if __has_builtin(__builtin_amdgcn_tanhf)
  return __builtin_amdgcn_tanhf(x);
#else
  return 1.0f - 2.0f * frcp_(1.0f + fexp_(2.0f * x));
#endif
}

extern "C" __global__ __launch_bounds__(THREADS, 1)
void lstm_reward_kernel(const int* __restrict__ tokens,
                        const float* __restrict__ emb,
                        const float* __restrict__ W_ih,
                        const float* __restrict__ W_hh,
                        const float* __restrict__ b_ih,
                        const float* __restrict__ b_hh,
                        const float* __restrict__ W1,
                        const float* __restrict__ b1,
                        const float* __restrict__ W2,
                        const float* __restrict__ b2,
                        float* __restrict__ out)
{
  extern __shared__ char smem[];
  _Float16* Wc    = (_Float16*)(smem + OFF_WC);    // combined [W_hh | W_ih] rows: [512][192]
  _Float16* W1l   = (_Float16*)(smem + OFF_W1);    // [64][128]
  float*    bl    = (float*)   (smem + OFF_BIAS);  // b_ih + b_hh : [512]
  _Float16* AsAll = (_Float16*)(smem + OFF_AS);    // per-wave [16][192] = (h || x_t)

  const int tid = threadIdx.x;

  // ---- one-time staging: f32 -> f16 weights into LDS (weights are L2-resident) ----
  for (int i = tid; i < NG * HH; i += THREADS) {
    int n = i >> 7, k = i & 127;
    Wc[n * KC + k] = (_Float16)W_hh[i];
  }
  for (int i = tid; i < NG * EE; i += THREADS) {
    int n = i >> 6, k = i & 63;
    Wc[n * KC + HH + k] = (_Float16)W_ih[i];
  }
  for (int i = tid; i < 64 * HH; i += THREADS) W1l[i] = (_Float16)W1[i];
  for (int i = tid; i < NG; i += THREADS) bl[i] = b_ih[i] + b_hh[i];
  __syncthreads();

  const int wave   = tid >> 5;          // wave32
  const int lane   = tid & 31;
  const int lr     = lane & 15;         // row (A) / col (B,D) within 16-tile
  const int hiHalf = (lane >= 16);
  const int rowBase = blockIdx.x * BLK_ROWS + wave * 16;
  _Float16* As = AsAll + wave * (16 * KC);

  // zero initial hidden state region of A-staging
  for (int i = lane; i < 16 * HH; i += 32) {
    int m = i >> 7, k = i & 127;
    As[m * KC + k] = (_Float16)0.0f;
  }

  const v8f vzero = {0.f, 0.f, 0.f, 0.f, 0.f, 0.f, 0.f, 0.f};
  v8f cst[8];                            // cell state tiles, D-layout, f32 (64 VGPRs)
  #pragma unroll
  for (int i = 0; i < 8; ++i) cst[i] = vzero;

  // ISA fragment addressing:
  //  A 16x32 f16: lanes 0-15 hold k {0..7,16..23}, lanes 16-31 hold k {8..15,24..31}
  //  B 32x16 f16: lanes 0-15 hold k {0..7,16..23}, lanes 16-31 hold k {8..15,24..31}
  const int aksel = hiHalf ? 8 : 0;
  const int bksel = hiHalf ? 8 : 0;

  #pragma unroll 1
  for (int t = 0; t < TT; ++t) {
    // ---- stage x_t = emb[tokens[:, t]] as f16 into As cols [128..192) ----
    {
      int r  = lane >> 1;                // 2 lanes per row
      int cb = (lane & 1) << 5;          // 0 or 32
      int tok = tokens[(rowBase + r) * TT + t];
      const float* ep = emb + tok * EE + cb;
      _Float16* dp = As + r * KC + HH + cb;
      #pragma unroll
      for (int c4 = 0; c4 < 32; c4 += 4) {
        F4 v = *(const F4*)(ep + c4);
        dp[c4 + 0] = (_Float16)v.x;
        dp[c4 + 1] = (_Float16)v.y;
        dp[c4 + 2] = (_Float16)v.z;
        dp[c4 + 3] = (_Float16)v.w;
      }
    }

    // ---- load 6 A-fragments of (h || x_t), K = 0..191 ----
    v16h a[6];
    #pragma unroll
    for (int kc = 0; kc < 6; ++kc) {
      const _Float16* p = As + lr * KC + kc * 32 + aksel;
      a[kc] = frag_load(p, p + 16);
    }

    // ---- gates = (h||x) @ [W_hh|W_ih]^T + bias ; 8 N-chunks x 4 gates ----
    #pragma unroll
    for (int nc = 0; nc < 8; ++nc) {
      v8f gi = vzero, gf = vzero, gg = vzero, go = vzero;
      #pragma unroll
      for (int kc = 0; kc < 6; ++kc) {
        const _Float16* pi = Wc + (0 * HH + nc * 16 + lr) * KC + kc * 32 + bksel;
        const _Float16* pf = Wc + (1 * HH + nc * 16 + lr) * KC + kc * 32 + bksel;
        const _Float16* pg = Wc + (2 * HH + nc * 16 + lr) * KC + kc * 32 + bksel;
        const _Float16* po = Wc + (3 * HH + nc * 16 + lr) * KC + kc * 32 + bksel;
        gi = wmma_f16(a[kc], frag_load(pi, pi + 16), gi);
        gf = wmma_f16(a[kc], frag_load(pf, pf + 16), gf);
        gg = wmma_f16(a[kc], frag_load(pg, pg + 16), gg);
        go = wmma_f16(a[kc], frag_load(po, po + 16), go);
      }
      float bia = bl[0 * HH + nc * 16 + lr];
      float bfa = bl[1 * HH + nc * 16 + lr];
      float bga = bl[2 * HH + nc * 16 + lr];
      float boa = bl[3 * HH + nc * 16 + lr];
      #pragma unroll
      for (int r = 0; r < 8; ++r) {
        float iv = fsig_(gi[r] + bia);
        float fv = fsig_(gf[r] + bfa);
        float gv = ftanh_(gg[r] + bga);
        float ov = fsig_(go[r] + boa);
        float cv = fv * cst[nc][r] + iv * gv;
        cst[nc][r] = cv;
        float hv = ov * ftanh_(cv);
        // write h_new back in D-layout position -> row-major f16 staging
        int mm = r + (hiHalf ? 8 : 0);
        As[mm * KC + nc * 16 + lr] = (_Float16)hv;
      }
    }
  }

  // ---- head: z = relu(h @ W1^T + b1); r = sigmoid(z @ W2^T + b2) ----
  v16h ha[4];
  #pragma unroll
  for (int kc = 0; kc < 4; ++kc) {
    const _Float16* p = As + lr * KC + kc * 32 + aksel;
    ha[kc] = frag_load(p, p + 16);
  }
  float pacc[8];
  #pragma unroll
  for (int r = 0; r < 8; ++r) pacc[r] = 0.f;

  #pragma unroll
  for (int nt = 0; nt < 4; ++nt) {
    v8f z = vzero;
    #pragma unroll
    for (int kc = 0; kc < 4; ++kc) {
      const _Float16* bp = W1l + (nt * 16 + lr) * HH + kc * 32 + bksel;
      z = wmma_f16(ha[kc], frag_load(bp, bp + 16), z);
    }
    float b1v = b1[nt * 16 + lr];
    float w2v = W2[nt * 16 + lr];
    #pragma unroll
    for (int r = 0; r < 8; ++r) {
      float zv = z[r] + b1v;
      zv = zv > 0.f ? zv : 0.f;
      pacc[r] += zv * w2v;
    }
  }

  const float b2v = b2[0];
  #pragma unroll
  for (int r = 0; r < 8; ++r) {
    float v = pacc[r];
    v += __shfl_xor(v, 1, 32);
    v += __shfl_xor(v, 2, 32);
    v += __shfl_xor(v, 4, 32);
    v += __shfl_xor(v, 8, 32);   // sum over the 16 N-lanes of each half
    if (lr == 0) {
      int mm = r + (hiHalf ? 8 : 0);
      out[rowBase + mm] = fsig_(v + b2v);
    }
  }
}

extern "C" void kernel_launch(void* const* d_in, const int* in_sizes, int n_in,
                              void* d_out, int out_size, void* d_ws, size_t ws_size,
                              hipStream_t stream) {
  (void)n_in; (void)d_ws; (void)ws_size; (void)out_size;
  const int*   tokens = (const int*)  d_in[0];
  const float* emb    = (const float*)d_in[1];
  const float* W_ih   = (const float*)d_in[2];
  const float* W_hh   = (const float*)d_in[3];
  const float* b_ih   = (const float*)d_in[4];
  const float* b_hh   = (const float*)d_in[5];
  const float* W1     = (const float*)d_in[6];
  const float* b1     = (const float*)d_in[7];
  const float* W2     = (const float*)d_in[8];
  const float* b2     = (const float*)d_in[9];
  float* out = (float*)d_out;

  const int Bsz    = in_sizes[0] / TT;        // 16384
  const int blocks = Bsz / BLK_ROWS;          // 256

  hipFuncSetAttribute((const void*)lstm_reward_kernel,
                      hipFuncAttributeMaxDynamicSharedMemorySize, (int)SMEM_BYTES);
  lstm_reward_kernel<<<blocks, THREADS, SMEM_BYTES, stream>>>(
      tokens, emb, W_ih, W_hh, b_ih, b_hh, W1, b1, W2, b2, out);
}
